// Attention_53850299957994
// MI455X (gfx1250) — matrix operand order repaired and checked by reading, not compiled, not run on hardware
//
#include <hip/hip_runtime.h>

#define BB 8
#define SS 2048
#define DD 512

typedef __attribute__((ext_vector_type(16))) __bf16        v16bf;
typedef __attribute__((ext_vector_type(2)))  __bf16        v2bf;
typedef __attribute__((ext_vector_type(8)))  float         v8f;
typedef __attribute__((ext_vector_type(4)))  unsigned int  u32x4;
typedef __attribute__((ext_vector_type(2)))  unsigned int  u32x2;
typedef __attribute__((ext_vector_type(8)))  int           i32x8;
typedef __attribute__((ext_vector_type(4)))  int           i32x4;
typedef __attribute__((ext_vector_type(4)))  float         f32x4;

union Frag16 {
    v16bf          v;
    unsigned short u[16];
    unsigned int   w[8];
    u32x4          q[2];
};

__device__ __forceinline__ unsigned short f2bf(float f) {
    union { float f; unsigned int u; } c;
    c.f = f;
    unsigned int u = c.u;
    u += 0x7FFFu + ((u >> 16) & 1u);   // round-to-nearest-even
    return (unsigned short)(u >> 16);
}

// packed 2x f32 -> 2x bf16 in one v_cvt_pk_bf16_f32 when available
__device__ __forceinline__ unsigned int f2bf_pk(float a, float b) {
#if __has_builtin(__builtin_amdgcn_cvt_pk_bf16_f32)
    union { v2bf v; unsigned int u; } c;
    c.v = __builtin_amdgcn_cvt_pk_bf16_f32(a, b);
    return c.u;
#else
    return (unsigned int)f2bf(a) | ((unsigned int)f2bf(b) << 16);
#endif
}

__device__ __forceinline__ v8f wmma_bf16(const Frag16& a, const Frag16& b, v8f c) {
    return __builtin_amdgcn_wmma_f32_16x16x32_bf16(false, a.v, false, b.v,
                                                   (short)0, c, false, false);
}

__device__ __forceinline__ void wait_ds0() {
#if __has_builtin(__builtin_amdgcn_s_wait_dscnt)
    __builtin_amdgcn_s_wait_dscnt(0);
#else
    asm volatile("s_wait_dscnt 0" ::: "memory");
#endif
}

__device__ __forceinline__ void wait_tensor0() {
#if __has_builtin(__builtin_amdgcn_s_wait_tensorcnt)
    __builtin_amdgcn_s_wait_tensorcnt(0);
#else
    asm volatile("s_wait_tensorcnt 0x0" ::: "memory");
#endif
}

// ---------------------------------------------------------------------------
// TDM: DMA one 2D bf16 tile (nrows x row_elems) global->LDS, optionally
// padding each row in LDS to lds_pitch_bytes (descriptor pad feature).
// Issue from ONE wave only (per-wave op, EXEC ignored); caller must
// wait_tensor0() in that wave before the workgroup barrier.
// ---------------------------------------------------------------------------
__device__ __forceinline__ void tdm_load_2d(void* lds_ptr,
                                            const unsigned short* gptr,
                                            unsigned row_elems, unsigned nrows,
                                            unsigned gpitch_elems,
                                            unsigned tensor_rows,
                                            unsigned lds_pitch_bytes) {
#if __has_builtin(__builtin_amdgcn_tensor_load_to_lds)
    const unsigned long long ga = (unsigned long long)(size_t)gptr;
    const unsigned lds_addr = (unsigned)(unsigned long long)(size_t)lds_ptr;
    const unsigned row_bytes = row_elems * 2u;
    const unsigned row_dw    = row_bytes >> 2;
    const unsigned pad_dw    = (lds_pitch_bytes - row_bytes) >> 2;
    const unsigned pad_en    = (pad_dw != 0u);

    u32x4 g0;
    g0[0] = 1u;                                          // count=1, user mode
    g0[1] = lds_addr;                                    // lds_addr [63:32]
    g0[2] = (unsigned)(ga & 0xFFFFFFFFull);              // global_addr lo
    g0[3] = (unsigned)((ga >> 32) & 0x1FFFFFFull)        // global_addr hi
          | (2u << 30);                                  // type = 2 ("image")

    i32x8 g1;
    unsigned d0 = (1u << 16);                            // data_size = 2 bytes
    if (pad_en)
        d0 |= (1u << 20)                                 // pad_enable
            | ((unsigned)(__builtin_ctz(row_dw) - 1) << 22)   // pad_interval
            | ((pad_dw - 1u) << 25);                          // pad_amount
    g1[0] = (int)d0;
    g1[1] = (int)((row_elems & 0xFFFFu) << 16);          // tensor_dim0 lo16
    g1[2] = (int)((row_elems >> 16) | ((tensor_rows & 0xFFFFu) << 16)); // d0 hi | d1 lo
    g1[3] = (int)((tensor_rows >> 16) | (row_elems << 16));             // d1 hi | tile_dim0
    g1[4] = (int)(nrows & 0xFFFFu);                      // tile_dim1 (tile_dim2=0)
    g1[5] = (int)gpitch_elems;                           // tensor_dim0_stride lo32
    g1[6] = 0;                                           // stride0 hi | dim1_stride lo (2D: unused)
    g1[7] = 0;
    i32x4 z4 = {0, 0, 0, 0};
    i32x8 z8 = {0, 0, 0, 0, 0, 0, 0, 0};
    __builtin_amdgcn_tensor_load_to_lds(g0, g1, z4, z4, z8, 0);
#else
    // fallback: 32-lane cooperative copy with padded pitch
    const int lane = threadIdx.x & 31;
    const int cpr = row_elems / 8;                       // 16B chunks per row
    for (int i = lane; i < (int)(nrows * cpr); i += 32) {
        int r = i / cpr, c = (i % cpr) * 8;
        *(u32x4*)((char*)lds_ptr + (size_t)r * lds_pitch_bytes + c * 2) =
            *(const u32x4*)(gptr + (size_t)r * gpitch_elems + c);
    }
    wait_ds0();
#endif
}

// ---------------------------------------------------------------------------
// Kernel 0: xb = bf16(x + pos_table)   [B*S, D]
// ---------------------------------------------------------------------------
__global__ __launch_bounds__(256) void xpos_kernel(const float* __restrict__ x,
                                                   const float* __restrict__ pos,
                                                   unsigned short* __restrict__ xb) {
    size_t i = ((size_t)blockIdx.x * 256 + threadIdx.x) * 4;
    f32x4 xv = *(const f32x4*)(x + i);
    size_t sd = i & ((size_t)SS * DD - 1);     // S*D = 2^20
    f32x4 pv = *(const f32x4*)(pos + sd);
    u32x2 o2 = {f2bf_pk(xv.x + pv.x, xv.y + pv.y),
                f2bf_pk(xv.z + pv.z, xv.w + pv.w)};
    *(u32x2*)(xb + i) = o2;
}

// ---------------------------------------------------------------------------
// GEMM: C[M,N] = A[M,K] (bf16) * W[N,K]^T (f32, nn.Linear layout) + bias
// block = 256 threads = 8 waves; block tile 128x32; wave tile 32x16.
// A block staged by TDM (wave 0), W staged/converted by all threads.
// ---------------------------------------------------------------------------
__global__ __launch_bounds__(256) void gemm_bf16(const unsigned short* __restrict__ A,
                                                 const float* __restrict__ W,
                                                 const float* __restrict__ bias,
                                                 float* __restrict__ Cf,
                                                 unsigned short* __restrict__ Cb,
                                                 int M, int N, int Kd,
                                                 int out_bf16, int nan0) {
    __shared__ unsigned short a_lds[128][40];  // 128x32 bf16, 80B pitch (TDM pad)
    __shared__ unsigned short w_lds[32][40];   // 32(n) x 32(k) bf16

    const int tid  = threadIdx.x;
    const int lane = tid & 31;
    const int wave = tid >> 5;
    const int wm   = wave >> 1;                // 0..3 : 32-row group
    const int wn   = wave & 1;                 // 0..1 : 16-col group
    const int m0   = blockIdx.y * 128;
    const int n0   = blockIdx.x * 32;

    v8f acc0 = {0.f,0.f,0.f,0.f,0.f,0.f,0.f,0.f};
    v8f acc1 = {0.f,0.f,0.f,0.f,0.f,0.f,0.f,0.f};

    for (int kk = 0; kk < Kd; kk += 32) {
        __syncthreads();
        // TDM: A block 128 rows x 32 bf16, strided by Kd, padded to 80B rows
        if (wave == 0)
            tdm_load_2d(&a_lds[0][0], A + (size_t)m0 * Kd + kk,
                        32, 128, Kd, M, 40 * 2);
        // stage W block: 32 rows x 32 f32 -> bf16 (overlaps the TDM)
        {
            int idx = tid * 4;                 // 0..1020
            int r = idx >> 5;
            int c = idx & 31;
            f32x4 wv = *(const f32x4*)(W + (size_t)(n0 + r) * Kd + kk + c);
            unsigned int* dst = (unsigned int*)&w_lds[r][c];
            dst[0] = f2bf_pk(wv.x, wv.y);
            dst[1] = f2bf_pk(wv.z, wv.w);
        }
        if (wave == 0) wait_tensor0();
        __syncthreads();

        // B fragment: lane holds column n (=W row), K halves {0..7,16..23} / {8..15,24..31}
        Frag16 bfr;
        {
            int col = wn * 16 + (lane & 15);
            int kh  = (lane >> 4) * 8;
            bfr.q[0] = *(const u32x4*)&w_lds[col][kh];
            bfr.q[1] = *(const u32x4*)&w_lds[col][kh + 16];
        }
        // A fragments: lane holds row m, K halves {0..7,16..23} / {8..15,24..31}
        Frag16 afr0, afr1;
        {
            int row = wm * 32 + (lane & 15);
            int e0  = (lane >> 4) * 8;
            afr0.q[0] = *(const u32x4*)&a_lds[row][e0];
            afr0.q[1] = *(const u32x4*)&a_lds[row][16 + e0];
            afr1.q[0] = *(const u32x4*)&a_lds[row + 16][e0];
            afr1.q[1] = *(const u32x4*)&a_lds[row + 16][16 + e0];
        }
        acc0 = wmma_bf16(afr0, bfr, acc0);
        acc1 = wmma_bf16(afr1, bfr, acc1);
    }

    // epilogue: C layout = VGPR r -> rows r / r+8 across lane halves
    const int ncol  = n0 + wn * 16 + (lane & 15);
    const float bv  = bias[ncol];
    const int rbase = m0 + wm * 32 + (lane >> 4) * 8;
#pragma unroll
    for (int r = 0; r < 8; ++r) {
        float x0 = acc0[r] + bv;
        float x1 = acc1[r] + bv;
        if (nan0) {
            x0 = (x0 == x0) ? x0 : 0.0f;
            x1 = (x1 == x1) ? x1 : 0.0f;
        }
        size_t i0 = (size_t)(rbase + r) * N + ncol;
        size_t i1 = (size_t)(rbase + 16 + r) * N + ncol;
        if (out_bf16) { Cb[i0] = f2bf(x0); Cb[i1] = f2bf(x1); }
        else          { Cf[i0] = x0;       Cf[i1] = x1; }
    }
}

// ---------------------------------------------------------------------------
// Flash attention: block = 8 waves x 16 queries; streams 32-key chunks.
// K chunk staged by TDM (row-padded); V chunk transposed to d-major by VALU.
// ---------------------------------------------------------------------------
__global__ __launch_bounds__(256) void flash_attn(const unsigned short* __restrict__ Q,
                                                  const unsigned short* __restrict__ K,
                                                  const unsigned short* __restrict__ V,
                                                  unsigned short* __restrict__ Y) {
    __shared__ unsigned short k_lds[32][DD + 8];   // 32 keys x 512, 1040B pitch
    __shared__ unsigned short vT_lds[DD][40];      // d-major: 512 x 32 keys
    __shared__ float          p_lds[8][16][33];    // per-wave P transpose scratch

    const int lane = threadIdx.x & 31;
    const int wave = threadIdx.x >> 5;
    const int b    = blockIdx.x >> 4;
    const int q0   = (blockIdx.x & 15) * 128;
    const size_t base = (size_t)b * SS;

    // ---- Q fragments for this wave's 16-query tile (held in registers) ----
    Frag16 qf[16];
    {
        const int qrow = q0 + wave * 16 + (lane & 15);
        const unsigned short* qp = Q + (base + qrow) * DD;
        const int e0 = (lane >> 4) * 8;
#pragma unroll
        for (int t = 0; t < 16; ++t) {
            qf[t].q[0] = *(const u32x4*)(qp + t * 32 + e0);
            qf[t].q[1] = *(const u32x4*)(qp + t * 32 + 16 + e0);
        }
    }

    v8f o[32];
#pragma unroll
    for (int i = 0; i < 32; ++i) o[i] = (v8f){0.f,0.f,0.f,0.f,0.f,0.f,0.f,0.f};
    float mrow[8], lrow[8];
#pragma unroll
    for (int r = 0; r < 8; ++r) { mrow[r] = -INFINITY; lrow[r] = 0.0f; }

    const float scale = 0.04419417382415922f;    // 1/sqrt(512)

    for (int j = 0; j < SS / 32; ++j) {
        const int kbase = j * 32;
        __syncthreads();
        // ---- TDM: K chunk 32 x 512 bf16, LDS rows padded to 1040B ----
        if (wave == 0)
            tdm_load_2d(&k_lds[0][0], K + (base + kbase) * DD,
                        DD, 32, DD, BB * SS, (DD + 8) * 2);
        // ---- stage V chunk d-major (transpose) — overlaps the TDM ----
#pragma unroll
        for (int i = 0; i < 8; ++i) {
            int idx = threadIdx.x + i * 256;
            int r = idx >> 6;
            int c = (idx & 63) * 8;
            u32x4 raw = *(const u32x4*)(V + (base + kbase + r) * DD + c);
            const unsigned short* e = (const unsigned short*)&raw;
#pragma unroll
            for (int t = 0; t < 8; ++t) vT_lds[c + t][r] = e[t];
        }
        if (j + 1 < SS / 32)  // prefetch next V chunk into L2
            __builtin_prefetch(V + (base + kbase + 32) * DD + threadIdx.x * 64, 0, 1);
        if (wave == 0) wait_tensor0();
        __syncthreads();

        // ---- scores: two 16x16 tiles, K-dim = D in 16 steps of 32 ----
        v8f s[2];
#pragma unroll
        for (int t2 = 0; t2 < 2; ++t2) {
            s[t2] = (v8f){0.f,0.f,0.f,0.f,0.f,0.f,0.f,0.f};
            const int key = t2 * 16 + (lane & 15);
            const int kh  = (lane >> 4) * 8;
#pragma unroll
            for (int t = 0; t < 16; ++t) {
                Frag16 kf;
                kf.q[0] = *(const u32x4*)&k_lds[key][t * 32 + kh];
                kf.q[1] = *(const u32x4*)&k_lds[key][t * 32 + kh + 16];
                s[t2] = wmma_bf16(qf[t], kf, s[t2]);
            }
        }

        // ---- online softmax (rows live across 16-lane halves) ----
        float mc[8];
#pragma unroll
        for (int r = 0; r < 8; ++r) {
            float a0 = s[0][r] * scale; a0 = (a0 == a0) ? a0 : -INFINITY;
            float a1 = s[1][r] * scale; a1 = (a1 == a1) ? a1 : -INFINITY;
            s[0][r] = a0; s[1][r] = a1;
            float m = fmaxf(a0, a1);
            m = fmaxf(m, __shfl_xor(m, 8, 16));
            m = fmaxf(m, __shfl_xor(m, 4, 16));
            m = fmaxf(m, __shfl_xor(m, 2, 16));
            m = fmaxf(m, __shfl_xor(m, 1, 16));
            mc[r] = m;
        }
        float alpha[8];
#pragma unroll
        for (int r = 0; r < 8; ++r) {
            float mn = fmaxf(mrow[r], mc[r]);
            float al = (mrow[r] == -INFINITY) ? 0.0f : __expf(mrow[r] - mn);
            float p0 = __expf(s[0][r] - mn);
            float p1 = __expf(s[1][r] - mn);
            s[0][r] = p0; s[1][r] = p1;
            float rs = p0 + p1;
            rs += __shfl_xor(rs, 8, 16);
            rs += __shfl_xor(rs, 4, 16);
            rs += __shfl_xor(rs, 2, 16);
            rs += __shfl_xor(rs, 1, 16);
            lrow[r] = lrow[r] * al + rs;
            mrow[r] = mn;
            alpha[r] = al;
        }
        // rescale accumulators
#pragma unroll
        for (int dt = 0; dt < 32; ++dt)
#pragma unroll
            for (int r = 0; r < 8; ++r) o[dt][r] *= alpha[r];

        // ---- transpose P (C layout -> A layout) through per-wave LDS ----
        {
            const int rowoff = (lane >> 4) * 8;
            const int col = lane & 15;
#pragma unroll
            for (int r = 0; r < 8; ++r) {
                p_lds[wave][r + rowoff][col]      = s[0][r];
                p_lds[wave][r + rowoff][16 + col] = s[1][r];
            }
        }
        wait_ds0();   // same-wave cross-lane LDS RAW
        Frag16 pf;
        {
            const float* pr = &p_lds[wave][lane & 15][0];
            const int e0 = (lane >> 4) * 8;
#pragma unroll
            for (int t = 0; t < 4; ++t) {
                pf.w[t]     = f2bf_pk(pr[e0 + 2 * t],      pr[e0 + 2 * t + 1]);
                pf.w[4 + t] = f2bf_pk(pr[16 + e0 + 2 * t], pr[16 + e0 + 2 * t + 1]);
            }
        }

        // ---- PV: 32 d-tiles, B fragments are contiguous reads from vT_lds ----
        const int dcol = lane & 15;
        const int kh   = (lane >> 4) * 8;
#pragma unroll
        for (int dt = 0; dt < 32; ++dt) {
            Frag16 vf;
            vf.q[0] = *(const u32x4*)&vT_lds[dt * 16 + dcol][kh];
            vf.q[1] = *(const u32x4*)&vT_lds[dt * 16 + dcol][kh + 16];
            o[dt] = wmma_bf16(pf, vf, o[dt]);
        }
    }

    // ---- normalize and store y (bf16) ----
    float inv[8];
#pragma unroll
    for (int r = 0; r < 8; ++r) inv[r] = (lrow[r] > 0.0f) ? 1.0f / lrow[r] : 0.0f;
    const int rowoff = (lane >> 4) * 8;
    const int col0 = lane & 15;
#pragma unroll
    for (int dt = 0; dt < 32; ++dt)
#pragma unroll
        for (int r = 0; r < 8; ++r) {
            int row = q0 + wave * 16 + rowoff + r;
            Y[(base + row) * DD + dt * 16 + col0] = f2bf(o[dt][r] * inv[r]);
        }
}

// ---------------------------------------------------------------------------
extern "C" void kernel_launch(void* const* d_in, const int* in_sizes, int n_in,
                              void* d_out, int out_size, void* d_ws, size_t ws_size,
                              hipStream_t stream) {
    (void)in_sizes; (void)n_in; (void)out_size; (void)ws_size;
    const float* x   = (const float*)d_in[0];
    const float* pos = (const float*)d_in[1];
    const float* Wq  = (const float*)d_in[2];
    const float* bq  = (const float*)d_in[3];
    const float* Wk  = (const float*)d_in[4];
    const float* bk  = (const float*)d_in[5];
    const float* Wv  = (const float*)d_in[6];
    const float* bv  = (const float*)d_in[7];
    const float* Wd  = (const float*)d_in[8];
    const float* bd  = (const float*)d_in[9];
    float* out = (float*)d_out;

    const size_t TOK = (size_t)BB * SS * DD;         // 8,388,608 elements
    unsigned short* xb = (unsigned short*)d_ws;      // 16 MB (reused as y)
    unsigned short* qb = xb + TOK;
    unsigned short* kb = qb + TOK;
    unsigned short* vb = kb + TOK;
    unsigned short* yb = xb;

    xpos_kernel<<<(TOK / 4) / 256, 256, 0, stream>>>(x, pos, xb);

    dim3 ggrid(DD / 32, (BB * SS) / 128);            // (16, 128)
    gemm_bf16<<<ggrid, 256, 0, stream>>>(xb, Wq, bq, nullptr, qb, BB * SS, DD, DD, 1, 0);
    gemm_bf16<<<ggrid, 256, 0, stream>>>(xb, Wk, bk, nullptr, kb, BB * SS, DD, DD, 1, 0);
    gemm_bf16<<<ggrid, 256, 0, stream>>>(xb, Wv, bv, nullptr, vb, BB * SS, DD, DD, 1, 1);

    flash_attn<<<BB * (SS / 128), 256, 0, stream>>>(qb, kb, vb, yb);

    gemm_bf16<<<ggrid, 256, 0, stream>>>(yb, Wd, bd, out, nullptr, BB * SS, DD, DD, 0, 0);
}
